// STRNN_16063177687565
// MI455X (gfx1250) — matrix operand order repaired and checked by reading, not compiled, not run on hardware
//
#include <hip/hip_runtime.h>
#include <hip/hip_bf16.h>

typedef __attribute__((ext_vector_type(16))) _Float16 v16h;
typedef __attribute__((ext_vector_type(8)))  _Float16 v8h;
typedef __attribute__((ext_vector_type(8)))  float    v8f;
typedef __attribute__((ext_vector_type(4)))  int      v4i;

typedef __attribute__((address_space(1))) v4i gv4i;   // global int4
typedef __attribute__((address_space(3))) v4i sv4i;   // LDS int4

#define VV 200000
#define II 512
#define HH 512
#define CC 4
#define NN 256
#define WW 32
#define PP 8
#define TT 4
#define H3 1536

#if __has_builtin(__builtin_amdgcn_global_load_async_to_lds_b128)
#define USE_ASYNC_LDS 1
#else
#define USE_ASYNC_LDS 0
#endif

__device__ __forceinline__ void wait_async0() {
#if __has_builtin(__builtin_amdgcn_s_wait_asynccnt)
  __builtin_amdgcn_s_wait_asynccnt(0);
#else
  asm volatile("s_wait_asynccnt 0" ::: "memory");
#endif
}

// Build the contiguous A-fragment for a vector living in LDS row `src`:
// lanes 0-15 hold K {k0..k0+15}; lanes 16-31 hold {k0+16..k0+31}.
// All lanes issue one 32-byte LDS load (broadcast), then mask.
__device__ __forceinline__ v16h vec_fragA(const _Float16* src, int k0, int hiK,
                                          bool active) {
  v16h a = *(const v16h*)(src + k0 + hiK);
  v16h z = {};
  return active ? a : z;
}

// ---------------------------------------------------------------------------
// Kernel 0: convert weights to f16 (w_ih, w_hh as-is; weight transposed so
// WMMA B-fragment loads are contiguous along K).
// ---------------------------------------------------------------------------
__global__ void conv_kernel(const float* __restrict__ w_ih,
                            const float* __restrict__ w_hh,
                            const float* __restrict__ weight,
                            _Float16* __restrict__ wih_h,
                            _Float16* __restrict__ whh_h,
                            _Float16* __restrict__ wT_h) {
  int idx = blockIdx.x * blockDim.x + threadIdx.x;
  const int nIH = H3 * II;
  const int nHH = H3 * HH;
  const int nW  = HH * HH;
  if (idx < nIH) {
    wih_h[idx] = (_Float16)w_ih[idx];
  } else if (idx < nIH + nHH) {
    int k = idx - nIH;
    whh_h[k] = (_Float16)w_hh[k];
  } else if (idx < nIH + nHH + nW) {
    int t = idx - nIH - nHH;
    int j = t >> 9, k = t & 511;           // wT[j][k] = weight[k][j]
    wT_h[t] = (_Float16)weight[k * HH + j];
  }
}

// ---------------------------------------------------------------------------
// Kernel 1: node_emb = mean over 32 word embeddings, stored f16 (WMMA A input)
// ---------------------------------------------------------------------------
__global__ void embed_mean_kernel(const int* __restrict__ x_index,
                                  const float* __restrict__ embed,
                                  _Float16* __restrict__ emb_h) {
  int n = blockIdx.x;
  int j = threadIdx.x;                     // 0..255
  float a0 = 0.f, a1 = 0.f;
  for (int w = 0; w < WW; ++w) {
    long long row = x_index[n * WW + w];
    const float* p = embed + row * II;
    a0 += p[j];
    a1 += p[j + 256];
  }
  emb_h[n * II + j]       = (_Float16)(a0 * (1.0f / WW));
  emb_h[n * II + j + 256] = (_Float16)(a1 * (1.0f / WW));
}

// ---------------------------------------------------------------------------
// Kernel 2: gi_all[m][n] = dot(node_emb[m], w_ih[n]) + b_ih[n]
// Full-utilization WMMA GEMM: M=256, N=1536, K=512. One wave per 16x16 tile.
// ---------------------------------------------------------------------------
__global__ void gemm_gi_kernel(const _Float16* __restrict__ emb_h,
                               const _Float16* __restrict__ wih_h,
                               const float* __restrict__ b_ih,
                               float* __restrict__ gi) {
  int wave = (blockIdx.x * blockDim.x + threadIdx.x) >> 5;   // 0..1535
  int lane = threadIdx.x & 31;
  int mt = wave / 96;                       // 0..15
  int nt = wave % 96;                       // 0..95
  int bm = mt * 16, bn = nt * 16;
  int r   = lane & 15;
  int hiK = (lane < 16) ? 0 : 8;            // unused; A also loads at bK
  int bK  = (lane < 16) ? 0 : 16;           // B K-chunk offset
  const _Float16* arow = emb_h + (bm + r) * II;
  const _Float16* brow = wih_h + (bn + r) * II;
  v8f c = {};
  for (int k0 = 0; k0 < II; k0 += 32) {
    v16h a = *(const v16h*)(arow + k0 + bK);
    v16h b = *(const v16h*)(brow + k0 + bK);
    c = __builtin_amdgcn_wmma_f32_16x16x32_f16(false, a, false, b, (short)0, c, false, false);
  }
  int col = bn + r;
  float bias = b_ih[col];
  int rowbase = bm + ((lane < 16) ? 0 : 8); // C/D layout: lanes 16-31 hold M=8..15
  #pragma unroll
  for (int rr = 0; rr < 8; ++rr)
    gi[(rowbase + rr) * H3 + col] = c[rr] + bias;
}

// ---------------------------------------------------------------------------
// Kernel 3: sequential scan. One persistent 1024-thread (32-wave) workgroup.
// Matvecs via WMMA: vector(s) in A rows 0/1, 16 weight rows as B columns.
// Phase B builds each A fragment once and feeds 3 accumulators (3 WMMAs/A).
// gi row for the node is prefetched to LDS with async-to-LDS (ASYNCcnt).
// ---------------------------------------------------------------------------
__global__ void __launch_bounds__(1024)
scan_kernel(const int* __restrict__ parents, const int* __restrict__ priors,
            const float* __restrict__ gi,
            const _Float16* __restrict__ whh_h, const _Float16* __restrict__ wT_h,
            const float* __restrict__ b_hh, const float* __restrict__ wproj,
            float* __restrict__ node_h) {
  __shared__ __attribute__((aligned(32))) _Float16 temph[HH];
  __shared__ __attribute__((aligned(32))) _Float16 topoh[HH];
  __shared__ __attribute__((aligned(32))) _Float16 h1h[HH];
  __shared__ float tempf[HH], topof[HH], h1f[HH];
  __shared__ float ghs[H3];
  __shared__ float us0[HH], us1[HH];
  __shared__ float red0[HH], red1[HH];
  __shared__ __attribute__((aligned(16))) float giL[H3];
  __shared__ float attS[2];

  int tid  = threadIdx.x;
  int lane = tid & 31;
  int wav  = tid >> 5;                       // 0..31
  int r    = lane & 15;
  int hiK  = (lane < 16) ? 0 : 8;
  int bK   = (lane < 16) ? 0 : 16;

  for (int n = 0; n < NN; ++n) {
#if USE_ASYNC_LDS
    // Async copy of gi[n] (6KB) to LDS; consumed in phase C, latency hidden
    // behind phases A+B. 384 lanes x 16B.
    if (tid < 384) {
      __builtin_amdgcn_global_load_async_to_lds_b128(
          (gv4i*)(gi + (size_t)n * H3 + tid * 4),
          (sv4i*)(giL + tid * 4),
          0, 0);
    }
#endif
    // ---- Phase A: temp/topo masked means (all referenced nodes are < n) ----
    if (tid < HH) {
      float ts = 0.f;
      for (int t = 0; t < TT; ++t) {
        int p = priors[n * TT + t];
        if (p >= 0) ts += node_h[p * HH + tid];
      }
      ts *= (1.0f / TT);
      tempf[tid] = ts; temph[tid] = (_Float16)ts;
      float tp = 0.f;
      for (int p = 0; p < PP; ++p) {
        int q = parents[n * PP + p];
        if (q >= 0) tp += node_h[q * HH + tid];
      }
      tp *= (1.0f / PP);
      topof[tid] = tp; topoh[tid] = (_Float16)tp;
    }
    __syncthreads();

    // ---- Phase B: gh = w_hh @ temp + b_hh via WMMA (96 tiles, 3/wave) ----
    {
      const _Float16* b0 = whh_h + ((wav      ) * 16 + r) * HH;
      const _Float16* b1 = whh_h + ((wav + 32 ) * 16 + r) * HH;
      const _Float16* b2 = whh_h + ((wav + 64 ) * 16 + r) * HH;
      v8f c0 = {}, c1 = {}, c2 = {};
      for (int k0 = 0; k0 < HH; k0 += 32) {
        v16h a = vec_fragA(temph, k0, bK, r == 0);   // A row 0 = temp
        v16h f0 = *(const v16h*)(b0 + k0 + bK);
        v16h f1 = *(const v16h*)(b1 + k0 + bK);
        v16h f2 = *(const v16h*)(b2 + k0 + bK);
        c0 = __builtin_amdgcn_wmma_f32_16x16x32_f16(false, a, false, f0, (short)0, c0, false, false);
        c1 = __builtin_amdgcn_wmma_f32_16x16x32_f16(false, a, false, f1, (short)0, c1, false, false);
        c2 = __builtin_amdgcn_wmma_f32_16x16x32_f16(false, a, false, f2, (short)0, c2, false, false);
      }
      if (lane < 16) {
        int i0 = (wav      ) * 16 + lane;
        int i1 = (wav + 32 ) * 16 + lane;
        int i2 = (wav + 64 ) * 16 + lane;
        ghs[i0] = c0[0] + b_hh[i0];                  // D[0][lane]
        ghs[i1] = c1[0] + b_hh[i1];
        ghs[i2] = c2[0] + b_hh[i2];
      }
    }
#if USE_ASYNC_LDS
    wait_async0();
#endif
    __syncthreads();

    // ---- Phase C: GRU gates ----
    if (tid < HH) {
#if USE_ASYNC_LDS
      float ir = giL[tid], iz = giL[HH + tid], in = giL[2 * HH + tid];
#else
      const float* g = gi + (size_t)n * H3;
      float ir = g[tid], iz = g[HH + tid], in = g[2 * HH + tid];
#endif
      float hr = ghs[tid], hz = ghs[HH + tid], hn = ghs[2 * HH + tid];
      float rr = 1.0f / (1.0f + __expf(-(ir + hr)));
      float zz = 1.0f / (1.0f + __expf(-(iz + hz)));
      float nn = tanhf(in + rr * hn);
      float h1 = (1.0f - zz) * nn + zz * tempf[tid];
      h1f[tid] = h1; h1h[tid] = (_Float16)h1;
    }
    __syncthreads();

    // ---- Phase D: u = tanh([topo; h1] @ weight) via WMMA (32 tiles, 1/wave)
    {
      int bj = wav * 16;
      const _Float16* brow = wT_h + (bj + r) * HH;
      v8f c = {};
      for (int k0 = 0; k0 < HH; k0 += 32) {
        const _Float16* src = (r == 1) ? h1h : topoh;  // row 0 topo, row 1 h1
        v16h a = vec_fragA(src, k0, bK, r < 2);
        v16h b = *(const v16h*)(brow + k0 + bK);
        c = __builtin_amdgcn_wmma_f32_16x16x32_f16(false, a, false, b, (short)0, c, false, false);
      }
      if (lane < 16) {
        us0[bj + lane] = tanhf(c[0]);         // D[0][lane] (topo row)
        us1[bj + lane] = tanhf(c[1]);         // D[1][lane] (h1 row)
      }
    }
    __syncthreads();

    // ---- Phase E: attention scores (dot with weight_proj) + softmax-of-2 ----
    if (tid < HH) {
      float wp = wproj[tid];
      red0[tid] = us0[tid] * wp;
      red1[tid] = us1[tid] * wp;
    }
    __syncthreads();
    for (int s = HH / 2; s > 0; s >>= 1) {
      if (tid < s) { red0[tid] += red0[tid + s]; red1[tid] += red1[tid + s]; }
      __syncthreads();
    }
    if (tid == 0) {
      float s0 = red0[0], s1 = red1[0];
      float m  = fmaxf(s0, s1);
      float e0 = __expf(s0 - m), e1 = __expf(s1 - m);
      float inv = 1.0f / (e0 + e1);
      attS[0] = e0 * inv; attS[1] = e1 * inv;
    }
    __syncthreads();

    // ---- Phase F: mix + commit hidden state ----
    if (tid < HH) {
      node_h[n * HH + tid] = attS[0] * topof[tid] + attS[1] * h1f[tid];
    }
    __threadfence();
    __syncthreads();
  }
}

// ---------------------------------------------------------------------------
// Kernel 4: logits = node_h[255] @ out_w^T + out_b; log_softmax -> d_out[0..3]
// ---------------------------------------------------------------------------
__global__ void logits_kernel(const float* __restrict__ node_h,
                              const float* __restrict__ out_w,
                              const float* __restrict__ out_b,
                              float* __restrict__ out) {
  __shared__ float lg[CC];
  int tid = threadIdx.x;
  int c = tid >> 5, lane = tid & 31;
  const float* h = node_h + (NN - 1) * HH;
  float acc = 0.f;
  for (int k = lane; k < HH; k += 32) acc += h[k] * out_w[c * HH + k];
  for (int off = 16; off > 0; off >>= 1) acc += __shfl_down(acc, off, 32);
  if (lane == 0) lg[c] = acc + out_b[c];
  __syncthreads();
  if (tid == 0) {
    float m = lg[0];
    for (int i = 1; i < CC; ++i) m = fmaxf(m, lg[i]);
    float s = 0.f;
    for (int i = 0; i < CC; ++i) s += __expf(lg[i] - m);
    float lse = __logf(s);
    for (int i = 0; i < CC; ++i) out[i] = lg[i] - m - lse;
  }
}

// ---------------------------------------------------------------------------
extern "C" void kernel_launch(void* const* d_in, const int* in_sizes, int n_in,
                              void* d_out, int out_size, void* d_ws, size_t ws_size,
                              hipStream_t stream) {
  const int*   x_index = (const int*)d_in[0];
  const int*   parents = (const int*)d_in[2];
  const int*   priors  = (const int*)d_in[3];
  const float* embed   = (const float*)d_in[4];
  const float* weight  = (const float*)d_in[5];
  const float* wproj   = (const float*)d_in[6];
  const float* w_ih    = (const float*)d_in[7];
  const float* w_hh    = (const float*)d_in[8];
  const float* b_ih    = (const float*)d_in[9];
  const float* b_hh    = (const float*)d_in[10];
  const float* out_w   = (const float*)d_in[11];
  const float* out_b   = (const float*)d_in[12];

  char* ws = (char*)d_ws;
  _Float16* wih_h = (_Float16*)ws; ws += (size_t)H3 * II * sizeof(_Float16);
  _Float16* whh_h = (_Float16*)ws; ws += (size_t)H3 * HH * sizeof(_Float16);
  _Float16* wT_h  = (_Float16*)ws; ws += (size_t)HH * HH * sizeof(_Float16);
  _Float16* emb_h = (_Float16*)ws; ws += (size_t)NN * II * sizeof(_Float16);
  float*    gi    = (float*)ws;    ws += (size_t)NN * H3 * sizeof(float);
  float*    nodeh = (float*)ws;    ws += (size_t)NN * HH * sizeof(float);

  int convN = H3 * II + H3 * HH + HH * HH;
  conv_kernel<<<(convN + 255) / 256, 256, 0, stream>>>(w_ih, w_hh, weight,
                                                       wih_h, whh_h, wT_h);
  embed_mean_kernel<<<NN, 256, 0, stream>>>(x_index, embed, emb_h);
  gemm_gi_kernel<<<384, 128, 0, stream>>>(emb_h, wih_h, b_ih, gi);   // 1536 waves = 1536 tiles
  scan_kernel<<<1, 1024, 0, stream>>>(parents, priors, gi, whh_h, wT_h,
                                      b_hh, wproj, nodeh);
  logits_kernel<<<1, 128, 0, stream>>>(nodeh, out_w, out_b, (float*)d_out);
}
